// KPCNN_89721866814098
// MI455X (gfx1250) — matrix operand order repaired and checked by reading, not compiled, not run on hardware
//
#include <hip/hip_runtime.h>
#include <math.h>

// ---------------------------------------------------------------------------
// KPCNN forward for gfx1250 (MI455X). All dense contractions go through
// v_wmma_f32_16x16x32_f16 (wave32). Each wave computes a 32x32 C tile
// (4 accumulators, 2 A frags x 2 B frags, 4 WMMAs per K-step) to halve
// A/B traffic and f32->f16 conversion work.
// ---------------------------------------------------------------------------

#define NKP       15
#define SLOPE     0.1f
#define BN_EPS    1e-5f
#define R2_GRAPH  (0.02f * 0.02f)
#define GRID_DL   0.2f
#define NCELL     5
#define MVOX      125           // 5^3
#define KN1       32
#define KN2       16
#define KNMAX     32
#define EXT1      (0.02f * 1.2f / 2.5f)
#define EXT2      (0.04f * 1.2f / 2.5f)
#define EXT3      (0.08f * 1.2f / 2.5f)
#define EXT4      (0.16f * 1.2f / 2.5f)

typedef __attribute__((ext_vector_type(16))) _Float16 v16h;
typedef __attribute__((ext_vector_type(8)))  float    v8f;
typedef __attribute__((ext_vector_type(4)))  float    f32x4;

// epilogue flags
#define EPI_BN         1
#define EPI_LRELU_PRE  2   // lrelu before residual add
#define EPI_LRELU_POST 4   // lrelu after residual add

// Branchless guarded load: always loads from a clamped, valid address,
// then selects zero -> v_cndmask, no exec-mask branches.
__device__ __forceinline__ float ldg_guard(const float* __restrict__ p,
                                           int r, int ld, int c,
                                           int Rmax, int Cmax) {
    int rc = r < Rmax ? r : Rmax - 1;
    int cc = c < Cmax ? c : Cmax - 1;
    float t = p[(size_t)rc * ld + cc];
    return (r < Rmax && c < Cmax) ? t : 0.f;
}

// ---- fragment builders ----------------------------------------------------
// A 16x32 f16 : lane -> M = lane&15 ; VGPR v -> K = (v&4?16:0)+8*half+2*(v&3)
//               per lane: two contiguous 8-float runs -> 4x float4 loads
// B 32x16 f16 : lane -> N = lane&15 ; VGPR v -> K = (v&4?16:0)+8*half+2*(v&3) (same as A)
//               per lane: 16 strided floats of one N column
// C 16x16 f32 : VGPR v -> M = v + 8*half ; N = lane&15
__device__ __forceinline__ v16h build_a_fast(const float* __restrict__ arow,
                                             int k0, int half) {
    const float* ab = arow + k0 + 8 * half;
    f32x4 a0 = *(const f32x4*)(ab);
    f32x4 a1 = *(const f32x4*)(ab + 4);
    f32x4 a2 = *(const f32x4*)(ab + 16);
    f32x4 a3 = *(const f32x4*)(ab + 20);
    v16h af;
#pragma unroll
    for (int i = 0; i < 4; ++i) {
        af[i]      = (_Float16)a0[i];
        af[i + 4]  = (_Float16)a1[i];
        af[i + 8]  = (_Float16)a2[i];
        af[i + 12] = (_Float16)a3[i];
    }
    return af;
}

__device__ __forceinline__ v16h build_b_fast(const float* __restrict__ bb, int ldb, int half) {
    v16h bf;
#pragma unroll
    for (int i = 0; i < 8; ++i) {
        bf[i]     = (_Float16)bb[(size_t)(8 * half + i) * ldb];
        bf[i + 8] = (_Float16)bb[(size_t)(16 + 8 * half + i) * ldb];
    }
    return bf;
}

__device__ __forceinline__ v16h build_a_guard(const float* __restrict__ A, int lda,
                                              int aRow, int k0, int half,
                                              int M, int K) {
    v16h af;
#pragma unroll
    for (int v = 0; v < 8; ++v) {
        int kk = k0 + ((v & 4) ? 16 : 0) + 8 * half + 2 * (v & 3);
        af[2 * v]     = (_Float16)ldg_guard(A, aRow, lda, kk,     M, K);
        af[2 * v + 1] = (_Float16)ldg_guard(A, aRow, lda, kk + 1, M, K);
    }
    return af;
}

__device__ __forceinline__ v16h build_b_guard(const float* __restrict__ Bm, int ldb,
                                              int k0, int half, int col, int K, int N) {
    v16h bf;
#pragma unroll
    for (int v = 0; v < 8; ++v) {
        int kk = k0 + ((v & 4) ? 16 : 0) + 8 * half + 2 * (v & 3);
        bf[2 * v]     = (_Float16)ldg_guard(Bm, kk,     ldb, col, K, N);
        bf[2 * v + 1] = (_Float16)ldg_guard(Bm, kk + 1, ldb, col, K, N);
    }
    return bf;
}

// Fused epilogue for one 16x16 quadrant at (r0, c0).
__device__ __forceinline__ void gemm_epilogue(
    v8f acc, int r0, int c0, int half, int l16,
    float* __restrict__ C, int ldc, int M, int N,
    const float* __restrict__ bn_g, const float* __restrict__ bn_b,
    const float* __restrict__ bn_m, const float* __restrict__ bn_v,
    const float* __restrict__ bias,
    const float* __restrict__ resid, int ldr, int flags)
{
    const int col = c0 + l16;
    if (col >= N) return;
    float scale = 1.f, shift = 0.f;
    if (flags & EPI_BN) {
        float iv = rsqrtf(bn_v[col] + BN_EPS);
        scale = bn_g[col] * iv;
        shift = bn_b[col] - bn_m[col] * scale;
    }
    const float bs = bias ? bias[col] : 0.f;
#pragma unroll
    for (int v = 0; v < 8; ++v) {
        int row = r0 + v + 8 * half;
        if (row >= M) continue;
        float val = acc[v];
        if (flags & EPI_BN)         val = val * scale + shift;
        if (flags & EPI_LRELU_PRE)  val = val > 0.f ? val : SLOPE * val;
        if (resid)                  val += resid[(size_t)row * ldr + col];
        if (flags & EPI_LRELU_POST) val = val > 0.f ? val : SLOPE * val;
        val += bs;
        C[(size_t)row * ldc + col] = val;
    }
}

// ---------------------------------------------------------------------------
// Generic WMMA GEMM:  C[M,N] = epilogue(A[M,K] * B[K,N])
// A,B row-major f32, converted to f16 on load. 32x32 tile / wave.
// Requires lda % 4 == 0 (host pads wf rows) and 16B-aligned base pointers.
// ---------------------------------------------------------------------------
__global__ __launch_bounds__(128) void wmma_gemm_kernel(
    const float* __restrict__ A, int lda,
    const float* __restrict__ Bm, int ldb,
    float* __restrict__ C, int ldc,
    int M, int N, int K,
    const float* __restrict__ bn_g, const float* __restrict__ bn_b,
    const float* __restrict__ bn_m, const float* __restrict__ bn_v,
    const float* __restrict__ bias,
    const float* __restrict__ resid, int ldr,
    int flags)
{
    const int wid  = threadIdx.x >> 5;
    const int lane = threadIdx.x & 31;
    const int tilesN = (N + 31) >> 5;          // 32-wide column strips
    const int tilesM = (M + 31) >> 5;          // 32-tall row strips
    const int nTiles = tilesM * tilesN;
    const int tile   = blockIdx.x * 4 + wid;   // wave-uniform
    if (tile >= nTiles) return;
    const int tm = tile / tilesN;
    const int tn = tile - tm * tilesN;
    const int half = lane >> 4;
    const int l16  = lane & 15;
    const int rowBase = tm * 32;
    const int colBase = tn * 32;

    if (rowBase + 32 <= M && colBase + 32 <= N) {
        // ---------------- interior fast path ----------------
        const float* __restrict__ arow0 = A + (size_t)(rowBase + l16) * lda;
        const float* __restrict__ arow1 = arow0 + (size_t)16 * lda;
        v8f acc00 = {}, acc01 = {}, acc10 = {}, acc11 = {};
        int k0 = 0;
        for (; k0 + 32 <= K; k0 += 32) {
            v16h af0 = build_a_fast(arow0, k0, half);
            v16h af1 = build_a_fast(arow1, k0, half);
            const float* bb = Bm + (size_t)k0 * ldb + colBase + l16;
            v16h bf0 = build_b_fast(bb, ldb, half);
            v16h bf1 = build_b_fast(bb + 16, ldb, half);
            acc00 = __builtin_amdgcn_wmma_f32_16x16x32_f16(false, af0, false, bf0, (short)0, acc00, false, false);
            acc01 = __builtin_amdgcn_wmma_f32_16x16x32_f16(false, af0, false, bf1, (short)0, acc01, false, false);
            acc10 = __builtin_amdgcn_wmma_f32_16x16x32_f16(false, af1, false, bf0, (short)0, acc10, false, false);
            acc11 = __builtin_amdgcn_wmma_f32_16x16x32_f16(false, af1, false, bf1, (short)0, acc11, false, false);
        }
        if (k0 < K) {   // K tail: rows/cols in range, only K guarded
            v16h af0 = build_a_guard(A, lda, rowBase + l16,      k0, half, M, K);
            v16h af1 = build_a_guard(A, lda, rowBase + 16 + l16, k0, half, M, K);
            v16h bf0 = build_b_guard(Bm, ldb, k0, half, colBase + l16,      K, N);
            v16h bf1 = build_b_guard(Bm, ldb, k0, half, colBase + 16 + l16, K, N);
            acc00 = __builtin_amdgcn_wmma_f32_16x16x32_f16(false, af0, false, bf0, (short)0, acc00, false, false);
            acc01 = __builtin_amdgcn_wmma_f32_16x16x32_f16(false, af0, false, bf1, (short)0, acc01, false, false);
            acc10 = __builtin_amdgcn_wmma_f32_16x16x32_f16(false, af1, false, bf0, (short)0, acc10, false, false);
            acc11 = __builtin_amdgcn_wmma_f32_16x16x32_f16(false, af1, false, bf1, (short)0, acc11, false, false);
        }
        gemm_epilogue(acc00, rowBase,      colBase,      half, l16, C, ldc, M, N,
                      bn_g, bn_b, bn_m, bn_v, bias, resid, ldr, flags);
        gemm_epilogue(acc01, rowBase,      colBase + 16, half, l16, C, ldc, M, N,
                      bn_g, bn_b, bn_m, bn_v, bias, resid, ldr, flags);
        gemm_epilogue(acc10, rowBase + 16, colBase,      half, l16, C, ldc, M, N,
                      bn_g, bn_b, bn_m, bn_v, bias, resid, ldr, flags);
        gemm_epilogue(acc11, rowBase + 16, colBase + 16, half, l16, C, ldc, M, N,
                      bn_g, bn_b, bn_m, bn_v, bias, resid, ldr, flags);
    } else {
        // ---------------- guarded edge path: per-16x16 quadrant ----------
#pragma unroll
        for (int qm = 0; qm < 2; ++qm) {
#pragma unroll
            for (int qn = 0; qn < 2; ++qn) {
                int r0 = rowBase + qm * 16;
                int c0 = colBase + qn * 16;
                if (r0 >= M || c0 >= N) continue;
                v8f acc = {};
                for (int k0 = 0; k0 < K; k0 += 32) {
                    v16h af = build_a_guard(A, lda, r0 + l16, k0, half, M, K);
                    v16h bf = build_b_guard(Bm, ldb, k0, half, c0 + l16, K, N);
                    acc = __builtin_amdgcn_wmma_f32_16x16x32_f16(
                        false, af, false, bf, (short)0, acc, false, false);
                }
                gemm_epilogue(acc, r0, c0, half, l16, C, ldc, M, N,
                              bn_g, bn_b, bn_m, bn_v, bias, resid, ldr, flags);
            }
        }
    }
}

// ---------------------------------------------------------------------------
// Fixed-radius KNN (top-kn smallest d2, self-loop included, ties -> low index)
// One thread per query point; candidate positions staged through LDS chunks.
// ---------------------------------------------------------------------------
__global__ __launch_bounds__(128) void knn_kernel(
    const float* __restrict__ pos,   // [B,P,3]
    const int*   __restrict__ valid, // [B,P] or null
    int P, int kn,
    int*   __restrict__ idx_out,     // [B,P,kn]
    float* __restrict__ mask_out)    // [B,P,kn]
{
    const int b = blockIdx.y;
    const int i = blockIdx.x * blockDim.x + threadIdx.x;
    const float* posb   = pos + (size_t)b * P * 3;
    const int*   validb = valid ? valid + (size_t)b * P : nullptr;

    float bd[KNMAX]; int bi[KNMAX];
#pragma unroll
    for (int q = 0; q < KNMAX; ++q) { bd[q] = 3.0e38f; bi[q] = 0; }

    float cx = 0.f, cy = 0.f, cz = 0.f;
    if (i < P) { cx = posb[i * 3 + 0]; cy = posb[i * 3 + 1]; cz = posb[i * 3 + 2]; }

    __shared__ float sx[256], sy[256], sz[256];
    __shared__ int   sv[256];
    for (int j0 = 0; j0 < P; j0 += 256) {
        int chunk = P - j0 < 256 ? P - j0 : 256;
        __syncthreads();
        for (int t = threadIdx.x; t < chunk; t += blockDim.x) {
            int j = j0 + t;
            sx[t] = posb[j * 3 + 0];
            sy[t] = posb[j * 3 + 1];
            sz[t] = posb[j * 3 + 2];
            sv[t] = validb ? validb[j] : 1;
        }
        __syncthreads();
        if (i < P) {
            for (int t = 0; t < chunk; ++t) {
                float dx = sx[t] - cx, dy = sy[t] - cy, dz = sz[t] - cz;
                float d2 = dx * dx + dy * dy + dz * dz;
                if (!sv[t]) d2 = 1.0e9f;
                if (d2 < bd[kn - 1]) {
                    int p = kn - 1;
                    while (p > 0 && bd[p - 1] > d2) {
                        bd[p] = bd[p - 1]; bi[p] = bi[p - 1]; --p;
                    }
                    bd[p] = d2; bi[p] = j0 + t;
                }
            }
        }
    }
    if (i < P) {
        int cval = validb ? validb[i] : 1;
        for (int q = 0; q < kn; ++q) {
            idx_out[((size_t)b * P + i) * kn + q]  = bi[q];
            mask_out[((size_t)b * P + i) * kn + q] =
                (cval && bd[q] <= R2_GRAPH) ? 1.0f : 0.0f;
        }
    }
}

// ---------------------------------------------------------------------------
// KPConv weighted-gather:  wf[n, k*Cin+c] = sum_j h[n,j,k] * feats[idx[n,j], c]
// with h = max(0, 1 - |rel - kp_k| / ext) * mask.  One block per point.
// Single-batch view: caller passes per-batch base pointers.
// ---------------------------------------------------------------------------
__global__ __launch_bounds__(256) void wfbuild_kernel(
    const float* __restrict__ pos,   // [P,3]
    const float* __restrict__ feats, // [P,Cin]
    int Cin,
    const int*   __restrict__ nidx,  // [P,kn]
    const float* __restrict__ nmask, // [P,kn]
    int kn,
    const float* __restrict__ kp,    // [NKP,3]
    float ext,
    float* __restrict__ wf,          // [P rows, row stride ldwf]
    int ldwf)
{
    const int n   = blockIdx.x;
    const int tid = threadIdx.x;

    __shared__ int   sidx[KNMAX];
    __shared__ float smask[KNMAX];
    __shared__ float srel[KNMAX][3];
    __shared__ float skp[NKP * 3];
    __shared__ float sh[KNMAX * NKP];
    __shared__ float sf[KNMAX * 256];   // kn*Cin <= 16*256 or 32*32

    if (tid < kn) {
        sidx[tid]  = nidx[(size_t)n * kn + tid];
        smask[tid] = nmask[(size_t)n * kn + tid];
    }
    if (tid < NKP * 3) skp[tid] = kp[tid];
    __syncthreads();

    const float cx = pos[n * 3 + 0], cy = pos[n * 3 + 1], cz = pos[n * 3 + 2];
    if (tid < kn) {
        int j = sidx[tid];
        srel[tid][0] = pos[j * 3 + 0] - cx;
        srel[tid][1] = pos[j * 3 + 1] - cy;
        srel[tid][2] = pos[j * 3 + 2] - cz;
    }
    __syncthreads();

    for (int t = tid; t < kn * NKP; t += blockDim.x) {
        int j = t / NKP, k = t - j * NKP;
        float dx = srel[j][0] - skp[k * 3 + 0];
        float dy = srel[j][1] - skp[k * 3 + 1];
        float dz = srel[j][2] - skp[k * 3 + 2];
        float d  = sqrtf(dx * dx + dy * dy + dz * dz);
        float h  = 1.f - d / ext;
        sh[j * NKP + k] = (h > 0.f ? h : 0.f) * smask[j];
    }
    for (int t = tid; t < kn * Cin; t += blockDim.x) {
        int j = t / Cin, c = t - j * Cin;
        sf[t] = feats[(size_t)sidx[j] * Cin + c];
    }
    __syncthreads();

    float* out = wf + (size_t)n * ldwf;
    for (int t = tid; t < NKP * Cin; t += blockDim.x) {
        int k = t / Cin, c = t - k * Cin;
        float acc = 0.f;
        for (int j = 0; j < kn; ++j)
            acc += sh[j * NKP + k] * sf[j * Cin + c];
        out[t] = acc;
    }
}

// ---------------------------------------------------------------------------
// Grid subsampling helpers
// ---------------------------------------------------------------------------
__global__ void zero_kernel(float* p, int n) {
    int i = blockIdx.x * blockDim.x + threadIdx.x;
    if (i < n) p[i] = 0.f;
}

__global__ void gs_scatter_kernel(
    const float* __restrict__ pos,  // [B,N,3]
    const float* __restrict__ x,    // [B,N,C]
    int total, int Nn, int C,
    float* __restrict__ possum,     // [B,MVOX,3]
    float* __restrict__ fsum,       // [B,MVOX,C]
    float* __restrict__ cnt)        // [B,MVOX]
{
    int gi = blockIdx.x * blockDim.x + threadIdx.x;
    if (gi >= total) return;
    int b = gi / Nn;
    float px = pos[(size_t)gi * 3 + 0];
    float py = pos[(size_t)gi * 3 + 1];
    float pz = pos[(size_t)gi * 3 + 2];
    int cxi = (int)floorf(px / GRID_DL); cxi = cxi < 0 ? 0 : (cxi > NCELL - 1 ? NCELL - 1 : cxi);
    int cyi = (int)floorf(py / GRID_DL); cyi = cyi < 0 ? 0 : (cyi > NCELL - 1 ? NCELL - 1 : cyi);
    int czi = (int)floorf(pz / GRID_DL); czi = czi < 0 ? 0 : (czi > NCELL - 1 ? NCELL - 1 : czi);
    int vid = (cxi * NCELL + cyi) * NCELL + czi;
    size_t vb = (size_t)b * MVOX + vid;
    atomicAdd(&cnt[vb], 1.f);
    atomicAdd(&possum[vb * 3 + 0], px);
    atomicAdd(&possum[vb * 3 + 1], py);
    atomicAdd(&possum[vb * 3 + 2], pz);
    const float* xr = x + (size_t)gi * C;
    float* fr = fsum + vb * C;
    for (int ch = 0; ch < C; ++ch) atomicAdd(&fr[ch], xr[ch]);
}

__global__ void gs_finalize_kernel(
    const float* __restrict__ possum, const float* __restrict__ fsum,
    const float* __restrict__ cnt, int total, int C,
    float* __restrict__ pos1, float* __restrict__ x1, int* __restrict__ valid1)
{
    int g = blockIdx.x * blockDim.x + threadIdx.x;
    if (g >= total) return;
    float c   = cnt[g];
    int   v   = c > 0.f ? 1 : 0;
    float inv = 1.f / fmaxf(c, 1.f);
    for (int d = 0; d < 3; ++d)
        pos1[g * 3 + d] = v ? possum[g * 3 + d] * inv : 1.0e6f;
    for (int ch = 0; ch < C; ++ch)
        x1[(size_t)g * C + ch] = fsum[(size_t)g * C + ch] * inv;
    valid1[g] = v;
}

// Masked average pool over valid voxels, then LeakyReLU.
__global__ __launch_bounds__(256) void pool_kernel(
    const float* __restrict__ x,     // [B,MVOX,C]
    const int*   __restrict__ valid, // [B,MVOX]
    int C, float* __restrict__ g)    // [B,C]
{
    int b = blockIdx.y;
    int c = blockIdx.x * blockDim.x + threadIdx.x;
    if (c >= C) return;
    float s = 0.f, w = 0.f;
    for (int m = 0; m < MVOX; ++m) {
        if (valid[b * MVOX + m]) {
            s += x[((size_t)b * MVOX + m) * C + c];
            w += 1.f;
        }
    }
    float val = s / fmaxf(w, 1.f);
    g[b * C + c] = val > 0.f ? val : SLOPE * val;
}

// ---------------------------------------------------------------------------
// Host side
// ---------------------------------------------------------------------------

// Input indices assuming JAX pytree flatten (dict keys sorted; 'W' < 'b..').
enum {
    IN_POINTS = 0, IN_FEATS = 1, IN_LENGTH = 2,
    B1_W = 3,  B1_BN = 4,  /* b,g,m,v */ B1_KP = 8,
    B2_W = 9,  B2_BN = 10, B2_DOWNW = 14, B2_DOWNBN = 15, B2_KP = 19,
    B2_RESW = 20, B2_RESBN = 21, B2_UPW = 25, B2_UPBN = 26,
    B3_W = 30, B3_BN = 31, B3_KP = 35,
    B3_RESW = 36, B3_RESBN = 37, B3_UPW = 41, B3_UPBN = 42,
    B4_W = 46, B4_BN = 47, B4_DOWNW = 51, B4_DOWNBN = 52, B4_KP = 56,
    B4_RESW = 57, B4_RESBN = 58, B4_UPW = 62, B4_UPBN = 63,
    OUT_W = 67, OUT_B = 68
};

extern "C" void kernel_launch(void* const* d_in, const int* in_sizes, int n_in,
                              void* d_out, int out_size, void* d_ws, size_t ws_size,
                              hipStream_t stream) {
    (void)n_in; (void)out_size; (void)ws_size;
    auto F = [&](int i) { return (const float*)d_in[i]; };

    const int Bb = in_sizes[IN_LENGTH];                 // 8
    const int Nn = in_sizes[IN_POINTS] / (Bb * 3);      // 4096
    const int NP2 = Bb * MVOX;                          // 1000 coarse points
    const int LDWF1 = 48;                               // padded 15*3 (=45) row

    const float* points = F(IN_POINTS);
    const float* feats  = F(IN_FEATS);

    // ---- workspace layout ----
    char* ws = (char*)d_ws;
    size_t off = 0;
    auto alloc = [&](size_t bytes) -> void* {
        void* p = ws + off;
        off = (off + bytes + 255) & ~(size_t)255;
        return p;
    };
    int*   idx1   = (int*)  alloc((size_t)Bb * Nn * KN1 * 4);
    float* mask1  = (float*)alloc((size_t)Bb * Nn * KN1 * 4);
    int*   idx2   = (int*)  alloc((size_t)NP2 * KN2 * 4);
    float* mask2  = (float*)alloc((size_t)NP2 * KN2 * 4);
    size_t wf_need = (size_t)NP2 * NKP * 256 * 4;               // b4 coarse
    size_t wf_alt  = (size_t)Nn * NKP * 32 * 4;                 // per-batch fine
    float* wfbuf  = (float*)alloc(wf_need > wf_alt ? wf_need : wf_alt);
    float* xA     = (float*)alloc((size_t)Bb * Nn * 64 * 4);    // b1 out
    float* xC     = (float*)alloc((size_t)Bb * Nn * 32 * 4);    // b2 down
    float* xD     = (float*)alloc((size_t)Bb * Nn * 32 * 4);    // b2 kpconv
    float* resbuf = (float*)alloc((size_t)Bb * Nn * 128 * 4);   // b2 residual
    float* xB     = (float*)alloc((size_t)Bb * Nn * 128 * 4);   // b2 out
    float* possum = (float*)alloc((size_t)NP2 * 3 * 4);
    float* fsum   = (float*)alloc((size_t)NP2 * 128 * 4);
    float* cnt    = (float*)alloc((size_t)NP2 * 4);
    float* pos1   = (float*)alloc((size_t)NP2 * 3 * 4);
    float* x1     = (float*)alloc((size_t)NP2 * 128 * 4);
    int*   valid1 = (int*)  alloc((size_t)NP2 * 4);
    float* t1     = (float*)alloc((size_t)NP2 * 128 * 4);
    float* r1     = (float*)alloc((size_t)NP2 * 512 * 4);
    float* x2     = (float*)alloc((size_t)NP2 * 512 * 4);
    float* t2     = (float*)alloc((size_t)NP2 * 256 * 4);
    float* t3     = (float*)alloc((size_t)NP2 * 256 * 4);
    float* r2     = (float*)alloc((size_t)NP2 * 1024 * 4);
    float* x3     = (float*)alloc((size_t)NP2 * 1024 * 4);
    float* gbuf   = (float*)alloc((size_t)Bb * 1024 * 4);

    auto run_gemm = [&](const float* A, int lda, const float* Bm, int ldb,
                        float* C, int ldc, int M, int N, int K,
                        int bn_base /* -1 for none */, const float* bias,
                        const float* resid, int ldr, int flags) {
        const float *g = nullptr, *bb = nullptr, *mm = nullptr, *vv = nullptr;
        if (bn_base >= 0) { bb = F(bn_base); g = F(bn_base + 1); mm = F(bn_base + 2); vv = F(bn_base + 3); }
        int tiles  = ((M + 31) / 32) * ((N + 31) / 32);
        int blocks = (tiles + 3) / 4;
        wmma_gemm_kernel<<<blocks, 128, 0, stream>>>(
            A, lda, Bm, ldb, C, ldc, M, N, K, g, bb, mm, vv, bias, resid, ldr, flags);
    };

    // ================= Stage 1: full resolution (P=4096, kn=32) ==========
    knn_kernel<<<dim3((Nn + 127) / 128, Bb), 128, 0, stream>>>(
        points, nullptr, Nn, KN1, idx1, mask1);

    // b1: SimpleBlock 3 -> 64   (wf rows padded to 48 so lda%4==0)
    for (int b = 0; b < Bb; ++b) {
        wfbuild_kernel<<<Nn, 256, 0, stream>>>(
            points + (size_t)b * Nn * 3, feats + (size_t)b * Nn * 3, 3,
            idx1 + (size_t)b * Nn * KN1, mask1 + (size_t)b * Nn * KN1, KN1,
            F(B1_KP), EXT1, wfbuf, LDWF1);
        run_gemm(wfbuf, LDWF1, F(B1_W), 64, xA + (size_t)b * Nn * 64, 64,
                 Nn, 64, NKP * 3, B1_BN, nullptr, nullptr, 0,
                 EPI_BN | EPI_LRELU_POST);
    }

    // b2: ResNet 64 -> 128 (bottleneck 32)
    run_gemm(xA, 64, F(B2_DOWNW), 32, xC, 32, Bb * Nn, 32, 64,
             B2_DOWNBN, nullptr, nullptr, 0, EPI_BN | EPI_LRELU_POST);
    for (int b = 0; b < Bb; ++b) {
        wfbuild_kernel<<<Nn, 256, 0, stream>>>(
            points + (size_t)b * Nn * 3, xC + (size_t)b * Nn * 32, 32,
            idx1 + (size_t)b * Nn * KN1, mask1 + (size_t)b * Nn * KN1, KN1,
            F(B2_KP), EXT2, wfbuf, NKP * 32);
        run_gemm(wfbuf, NKP * 32, F(B2_W), 32, xD + (size_t)b * Nn * 32, 32,
                 Nn, 32, NKP * 32, B2_BN, nullptr, nullptr, 0,
                 EPI_BN | EPI_LRELU_POST);
    }
    run_gemm(xA, 64, F(B2_RESW), 128, resbuf, 128, Bb * Nn, 128, 64,
             B2_RESBN, nullptr, nullptr, 0, EPI_BN);
    run_gemm(xD, 32, F(B2_UPW), 128, xB, 128, Bb * Nn, 128, 32,
             B2_UPBN, nullptr, resbuf, 128,
             EPI_BN | EPI_LRELU_PRE | EPI_LRELU_POST);

    // ================= Grid subsampling (0.2) =============================
    zero_kernel<<<(NP2 * 3 + 255) / 256, 256, 0, stream>>>(possum, NP2 * 3);
    zero_kernel<<<(NP2 * 128 + 255) / 256, 256, 0, stream>>>(fsum, NP2 * 128);
    zero_kernel<<<(NP2 + 255) / 256, 256, 0, stream>>>(cnt, NP2);
    gs_scatter_kernel<<<(Bb * Nn + 255) / 256, 256, 0, stream>>>(
        points, xB, Bb * Nn, Nn, 128, possum, fsum, cnt);
    gs_finalize_kernel<<<(NP2 + 255) / 256, 256, 0, stream>>>(
        possum, fsum, cnt, NP2, 128, pos1, x1, valid1);

    // ================= Stage 2: coarse (P=125, kn=16) =====================
    knn_kernel<<<dim3(1, Bb), 128, 0, stream>>>(
        pos1, valid1, MVOX, KN2, idx2, mask2);

    // b3: ResNet 128 -> 512 (no down)
    for (int b = 0; b < Bb; ++b)
        wfbuild_kernel<<<MVOX, 256, 0, stream>>>(
            pos1 + (size_t)b * MVOX * 3, x1 + (size_t)b * MVOX * 128, 128,
            idx2 + (size_t)b * MVOX * KN2, mask2 + (size_t)b * MVOX * KN2, KN2,
            F(B3_KP), EXT3, wfbuf + (size_t)b * MVOX * NKP * 128, NKP * 128);
    run_gemm(wfbuf, NKP * 128, F(B3_W), 128, t1, 128, NP2, 128, NKP * 128,
             B3_BN, nullptr, nullptr, 0, EPI_BN | EPI_LRELU_POST);
    run_gemm(x1, 128, F(B3_RESW), 512, r1, 512, NP2, 512, 128,
             B3_RESBN, nullptr, nullptr, 0, EPI_BN);
    run_gemm(t1, 128, F(B3_UPW), 512, x2, 512, NP2, 512, 128,
             B3_UPBN, nullptr, r1, 512,
             EPI_BN | EPI_LRELU_PRE | EPI_LRELU_POST);

    // b4: ResNet 512 -> 1024 (bottleneck 256)
    run_gemm(x2, 512, F(B4_DOWNW), 256, t2, 256, NP2, 256, 512,
             B4_DOWNBN, nullptr, nullptr, 0, EPI_BN | EPI_LRELU_POST);
    for (int b = 0; b < Bb; ++b)
        wfbuild_kernel<<<MVOX, 256, 0, stream>>>(
            pos1 + (size_t)b * MVOX * 3, t2 + (size_t)b * MVOX * 256, 256,
            idx2 + (size_t)b * MVOX * KN2, mask2 + (size_t)b * MVOX * KN2, KN2,
            F(B4_KP), EXT4, wfbuf + (size_t)b * MVOX * NKP * 256, NKP * 256);
    run_gemm(wfbuf, NKP * 256, F(B4_W), 256, t3, 256, NP2, 256, NKP * 256,
             B4_BN, nullptr, nullptr, 0, EPI_BN | EPI_LRELU_POST);
    run_gemm(x2, 512, F(B4_RESW), 1024, r2, 1024, NP2, 1024, 512,
             B4_RESBN, nullptr, nullptr, 0, EPI_BN);
    run_gemm(t3, 256, F(B4_UPW), 1024, x3, 1024, NP2, 1024, 256,
             B4_UPBN, nullptr, r2, 1024,
             EPI_BN | EPI_LRELU_PRE | EPI_LRELU_POST);

    // ================= Pool + classifier ==================================
    pool_kernel<<<dim3((1024 + 255) / 256, Bb), 256, 0, stream>>>(
        x3, valid1, 1024, gbuf);
    run_gemm(gbuf, 1024, F(OUT_W), 40, (float*)d_out, 40, Bb, 40, 1024,
             -1, F(OUT_B), nullptr, 0, 0);
}
